// SparseLayer_27831388078549
// MI455X (gfx1250) — matrix operand order repaired and checked by reading, not compiled, not run on hardware
//
#include <hip/hip_runtime.h>
#include <hip/hip_bf16.h>

#define N_POST  50000
#define N_IN    17400
#define NNZ_    1000000
#define N_BASIS 5
#define N_TYPES 10
#define T_TOK   64

typedef __attribute__((ext_vector_type(16))) _Float16 v16h;
typedef __attribute__((ext_vector_type(8)))  float    v8f;

// ---------------------------------------------------------------- zero fill
__global__ void zero_f4_kernel(float4* __restrict__ p, long long n4) {
    long long i = (long long)blockIdx.x * blockDim.x + threadIdx.x;
    if (i < n4) p[i] = make_float4(0.f, 0.f, 0.f, 0.f);
}

// ------------------------------------------------- xT[col*64 + t] = x[t, col]
__global__ void transpose_x_kernel(const float* __restrict__ x,
                                   float* __restrict__ xT) {
    int gid = blockIdx.x * blockDim.x + threadIdx.x;
    if (gid >= N_IN * T_TOK) return;
    int col = gid >> 6;
    int t   = gid & 63;
    xT[gid] = x[(size_t)t * N_IN + col];
}

// ---------------- stage 1: y[post, syn, t] += w[e] * xT[col_e, t]  (64 t/nnz)
__global__ void stage1_scatter_kernel(const int*   __restrict__ indices,  // [NNZ][2]
                                      const float* __restrict__ wts,      // [NNZ]
                                      const int*   __restrict__ syn_ids,  // [NNZ]
                                      const float* __restrict__ xT,       // [N_IN][64]
                                      float*       __restrict__ y) {      // [N_POST][10][64]
    long long gid = (long long)blockIdx.x * blockDim.x + threadIdx.x;
    int e = (int)(gid >> 6);
    int t = (int)(gid & 63);
    if (e >= NNZ_) return;
    int row = indices[2 * e];
    int col = indices[2 * e + 1];
    int syn = syn_ids[e];
    float v = wts[e] * xT[(size_t)col * T_TOK + t];
    unsafeAtomicAdd(&y[((size_t)row * N_TYPES + syn) * T_TOK + t], v);
}

// ---------------- stage 2: out[t, post, r] = sum_k y[post,k,t] * S[k,r]
// One wave = one (post, 16-token tile). M=16 tokens, K=10 (pad->32), N=5 (pad->16).
// f32 accuracy recovered with split-f16 3-product trick into the f32 accumulator.
__global__ void stage2_wmma_kernel(const float* __restrict__ y,   // [N_POST][10][64]
                                   const float* __restrict__ s,   // [10][5]
                                   float*       __restrict__ out) // [64][N_POST*5]
{
    int wave = blockIdx.x * (blockDim.x >> 5) + ((int)threadIdx.x >> 5);
    int lane = (int)threadIdx.x & 31;
    int post = wave >> 2;
    int t0   = (wave & 3) << 4;
    if (post >= N_POST) return;           // wave-uniform: EXEC stays all-ones
    int  m  = lane & 15;
    bool hi = lane >= 16;

    // A (16x32 f16, M=token, K=syn):  lane<16 element e -> K=e
    //                                 lane>=16 element e -> K=16+e (zero)
    const float* yp = y + (size_t)post * (N_TYPES * T_TOK) + (t0 + m);
    float af[16];
#pragma unroll
    for (int e = 0; e < 16; ++e) af[e] = 0.f;
    if (!hi) {
#pragma unroll
        for (int k = 0; k < N_TYPES; ++k) af[k] = yp[k * T_TOK];   // K = 0..9
    }

    // B (32x16 f16, K=syn, N=basis): lane<16 element e -> B[K=e, N=lane]; lane>=16 -> K>=16 (zero)
    float bf[16];
#pragma unroll
    for (int e = 0; e < 16; ++e) bf[e] = 0.f;
    if (!hi && m < N_BASIS) {
#pragma unroll
        for (int k = 0; k < N_TYPES; ++k) bf[k] = s[k * N_BASIS + m];
    }

    v16h a_hi, a_lo, b_hi, b_lo;
#pragma unroll
    for (int e = 0; e < 16; ++e) {
        _Float16 h = (_Float16)af[e];
        a_hi[e] = h;
        a_lo[e] = (_Float16)(af[e] - (float)h);
        _Float16 g = (_Float16)bf[e];
        b_hi[e] = g;
        b_lo[e] = (_Float16)(bf[e] - (float)g);
    }

    v8f c = {};
    c = __builtin_amdgcn_wmma_f32_16x16x32_f16(false, a_lo, false, b_hi, (short)0, c, false, false);
    c = __builtin_amdgcn_wmma_f32_16x16x32_f16(false, a_hi, false, b_lo, (short)0, c, false, false);
    c = __builtin_amdgcn_wmma_f32_16x16x32_f16(false, a_hi, false, b_hi, (short)0, c, false, false);

    // D 16x16 f32: element i -> (M = i + (hi?8:0), N = m). Store only N < 5.
    if (m < N_BASIS) {
        size_t cbase = (size_t)post * N_BASIS + m;
        int moff = hi ? 8 : 0;
#pragma unroll
        for (int i = 0; i < 8; ++i)
            out[(size_t)(t0 + moff + i) * ((size_t)N_POST * N_BASIS) + cbase] = c[i];
    }
}

// ---------------- fallback (no workspace): direct 5-way atomic scatter
__global__ void fallback_kernel(const int*   __restrict__ indices,
                                const float* __restrict__ wts,
                                const float* __restrict__ s,
                                const int*   __restrict__ syn_ids,
                                const float* __restrict__ x,
                                float*       __restrict__ out) {
    long long gid = (long long)blockIdx.x * blockDim.x + threadIdx.x;
    int e = (int)(gid >> 6);
    int t = (int)(gid & 63);
    if (e >= NNZ_) return;
    int row = indices[2 * e];
    int col = indices[2 * e + 1];
    int syn = syn_ids[e];
    float g = x[(size_t)t * N_IN + col] * wts[e];
    size_t base = (size_t)t * ((size_t)N_POST * N_BASIS) + (size_t)row * N_BASIS;
#pragma unroll
    for (int r = 0; r < N_BASIS; ++r)
        unsafeAtomicAdd(&out[base + r], g * s[syn * N_BASIS + r]);
}

extern "C" void kernel_launch(void* const* d_in, const int* in_sizes, int n_in_args,
                              void* d_out, int out_size, void* d_ws, size_t ws_size,
                              hipStream_t stream) {
    const float* x       = (const float*)d_in[0];  // (1,64,17400) f32
    const int*   indices = (const int*)d_in[1];    // (NNZ,2)
    const float* iw      = (const float*)d_in[2];  // (NNZ,)
    const float* sw      = (const float*)d_in[3];  // (10,5)
    const int*   syn     = (const int*)d_in[4];    // (NNZ,)
    float*       out     = (float*)d_out;          // 64*50000*5 f32

    const size_t y_elems  = (size_t)N_POST * N_TYPES * T_TOK;  // 32M floats
    const size_t y_bytes  = y_elems * sizeof(float);           // 128 MB
    const size_t xT_bytes = (size_t)N_IN * T_TOK * sizeof(float);

    const long long s1_threads = (long long)NNZ_ * T_TOK;      // 64M

    if (ws_size >= y_bytes + xT_bytes) {
        float* y  = (float*)d_ws;
        float* xT = (float*)((char*)d_ws + y_bytes);

        long long n4 = (long long)(y_elems / 4);
        zero_f4_kernel<<<(unsigned)((n4 + 255) / 256), 256, 0, stream>>>((float4*)y, n4);

        transpose_x_kernel<<<(N_IN * T_TOK + 255) / 256, 256, 0, stream>>>(x, xT);

        stage1_scatter_kernel<<<(unsigned)((s1_threads + 255) / 256), 256, 0, stream>>>(
            indices, iw, syn, xT, y);

        int waves = N_POST * 4;   // 4 token tiles per post, 8 waves per 256-thread block
        stage2_wmma_kernel<<<(waves + 7) / 8, 256, 0, stream>>>(y, sw, out);
    } else {
        long long o4 = (long long)out_size / 4;
        zero_f4_kernel<<<(unsigned)((o4 + 255) / 256), 256, 0, stream>>>((float4*)out, o4);
        fallback_kernel<<<(unsigned)((s1_threads + 255) / 256), 256, 0, stream>>>(
            indices, iw, sw, syn, x, out);
    }
}
